// KNNC_84516366450717
// MI455X (gfx1250) — compile-verified, matrix operand order checked by reading, not run on hardware
//
#include <hip/hip_runtime.h>
#include <stdint.h>

// ---------------------------------------------------------------------------
// KNN classification (B=2048, P=50000, C=100, K=16) for MI455X / gfx1250.
// Roofline: 400 MB of distances streamed once -> ~17us floor at 23.3 TB/s.
// - Barrier-free double-buffered async global->LDS streaming per wave
//   (global_load_async_to_lds_b128 + s_wait_asynccnt), lanes consume only
//   bytes their own async loads wrote.
// - Wave-cooperative filter top-16: shared threshold tau + 64-slot LDS
//   candidate buffer per wave; compaction (rare) is a single __noinline__
//   routine doing 16 rounds of wave32 u64-key shfl-min extraction.
// - WMMA f32_16x16x32_f16 vote-sum epilogue, K-layout-invariant (A uniform
//   = 1/32, each neighbor row appears twice among the 32 K slots).
// ---------------------------------------------------------------------------

typedef __attribute__((ext_vector_type(16))) _Float16 v16h;
typedef __attribute__((ext_vector_type(8)))  float    v8f;

// exact types for the async builtin (param0 is AS(1) ptr to 16B int vector)
typedef int v4i_gcc __attribute__((__vector_size__(16)));
typedef __attribute__((address_space(1))) v4i_gcc* gp128_t;
typedef __attribute__((address_space(3))) v4i_gcc* lp128_t;
typedef __attribute__((address_space(3))) unsigned long long* lds_u64p;

#define KNN_K   16
#define BLK     128
#define CHUNKF  4096                 // floats per LDS chunk (16 KB)
#define SUBS    (CHUNKF / (BLK * 4)) // = 8 b128 async loads / thread / chunk

__device__ __forceinline__ void async_b128(const float* g, float* l) {
#if __has_builtin(__builtin_amdgcn_global_load_async_to_lds_b128)
  __builtin_amdgcn_global_load_async_to_lds_b128(
      (gp128_t)(void*)g, (lp128_t)(void*)l, 0, 0);
#else
  unsigned loff = (unsigned)(unsigned long long)
      ( (__attribute__((address_space(3))) char*) l );
  asm volatile("global_load_async_to_lds_b128 %0, %1, off"
               :: "v"(loff), "v"(g) : "memory");
#endif
}

__device__ __forceinline__ void wait_async_8() {
#if __has_builtin(__builtin_amdgcn_s_wait_asynccnt)
  __builtin_amdgcn_s_wait_asynccnt(8);
#else
  asm volatile("s_wait_asynccnt 8" ::: "memory");
#endif
  asm volatile("" ::: "memory"); // keep LDS reads below the wait
}

__device__ __forceinline__ void wait_async_0() {
#if __has_builtin(__builtin_amdgcn_s_wait_asynccnt)
  __builtin_amdgcn_s_wait_asynccnt(0);
#else
  asm volatile("s_wait_asynccnt 0" ::: "memory");
#endif
  asm volatile("" ::: "memory");
}

__device__ __forceinline__ unsigned long long u64min(unsigned long long a,
                                                     unsigned long long b) {
  return a < b ? a : b;
}

// Rare path, kept OUT of line so the ~170 DS-swizzle ops exist once, not at
// every consider() site. Reduces cand[0..count) to its 16 smallest u64 keys
// (left in slots 0..15) and returns the new threshold tau (16th value).
// Must be called by a full wave (EXEC all ones); count in (16, 64].
__device__ __attribute__((noinline)) float knn_compact(lds_u64p candw,
                                                       int lane, int count) {
  __builtin_amdgcn_wave_barrier();
  unsigned long long a = candw[lane];
  unsigned long long q = candw[lane + 32];
  if (lane      >= count) a = ~0ULL;
  if (lane + 32 >= count) q = ~0ULL;
  unsigned long long keep = ~0ULL;
  unsigned long long last = 0;
#pragma unroll
  for (int r = 0; r < KNN_K; ++r) {
    unsigned long long m = u64min(a, q);
    unsigned long long g = m;
#pragma unroll
    for (int off = 16; off >= 1; off >>= 1)
      g = u64min(g, (unsigned long long)__shfl_xor(g, off, 32));
    if (a == g) a = ~0ULL; else if (q == g) q = ~0ULL;  // keys are unique
    if (lane == r) keep = g;
    last = g;
  }
  if (lane < KNN_K) candw[lane] = keep;
  __builtin_amdgcn_wave_barrier();
  return __uint_as_float((unsigned)(last >> 32));
}

__global__ __launch_bounds__(BLK) void knnc_kernel(
    const float* __restrict__ dist,    // [B, P]
    const float* __restrict__ labels,  // [P, C] one-hot f32
    float* __restrict__ out,           // [B, C]
    int P, int C) {
  const int b    = blockIdx.x;
  const int tid  = threadIdx.x;
  const int lane = tid & 31;
  const int wave = tid >> 5;
  const float* row = dist + (size_t)b * (size_t)P;

  __shared__ __align__(16) float sbuf[2][CHUNKF];  // 32 KB double buffer
  __shared__ unsigned long long cand[4][64];       // per-wave candidate pool
  __shared__ int sfidx[KNN_K];                     // final 16 indices

  // ---- wave-cooperative filter top-16 state (wave-uniform) ---------------
  float tau  = __builtin_inff();  // upper bound on wave's 16th smallest
  int   count = 0;                // valid slots in cand[wave]
  const unsigned lanemask_lt = (1u << lane) - 1u;

  // must be called by the FULL wave (count/tau stay wave-uniform).
  // check=false: caller guarantees gi < P for every lane.
  auto consider = [&](float x, int gi, bool check) {
    const bool pred = (x < tau) && (!check || gi < P);
    const unsigned mask = (unsigned)__ballot(pred);
    if (mask) {
      const int ofs = __popc(mask & lanemask_lt);
      if (pred)
        cand[wave][count + ofs] =
            ((unsigned long long)__float_as_uint(x) << 32) | (unsigned)gi;
      count += __popc(mask);
      if (count > 32) {
        tau = knn_compact((lds_u64p)&cand[wave][0], lane, count);
        count = KNN_K;
      }
    }
  };

  auto issue = [&](int ch, int buf) {
    const int base = ch * CHUNKF;
#pragma unroll
    for (int s = 0; s < SUBS; ++s) {
      const int slot = s * BLK + tid;
      const int fi   = base + slot * 4;
      const int safe = (fi + 4 <= P) ? fi : 0;  // clamp OOB tail to row start
      async_b128(row + safe, &sbuf[buf][slot * 4]);
    }
  };

  const int nch = (P + CHUNKF - 1) / CHUNKF;

  // Barrier-free async pipeline: each lane consumes exactly the LDS bytes its
  // own async loads wrote; per-wave ASYNCcnt ordering is the only sync needed.
  issue(0, 0);
  for (int c = 0; c < nch; ++c) {
    if (c + 1 < nch) { issue(c + 1, (c + 1) & 1); wait_async_8(); }
    else             { wait_async_0(); }
    const int base = c * CHUNKF;
    const float4* bufv = reinterpret_cast<const float4*>(sbuf[c & 1]);
    if (c + 1 < nch) {
      // full chunk: every gi < P, no range compare in the hot path
#pragma unroll
      for (int s = 0; s < SUBS; ++s) {
        const int fi = base + (s * BLK + tid) * 4;
        float4 q = bufv[s * BLK + tid];
        consider(q.x, fi,     false); consider(q.y, fi + 1, false);
        consider(q.z, fi + 2, false); consider(q.w, fi + 3, false);
      }
    } else {
      // ragged last chunk: per-element validity via gi < P
#pragma unroll
      for (int s = 0; s < SUBS; ++s) {
        const int fi = base + (s * BLK + tid) * 4;
        float4 q = bufv[s * BLK + tid];  // garbage if OOB; filtered by gi < P
        consider(q.x, fi,     true); consider(q.y, fi + 1, true);
        consider(q.z, fi + 2, true); consider(q.w, fi + 3, true);
      }
    }
  }
  // tail elements (P % 4): full-wave call, invalid lanes pass gi = P.
  {
    const int base4 = P & ~3;
    const int rem   = P & 3;
    float tx = 0.0f;
    int   tgi = P;
    if (tid < rem) { tx = row[base4 + tid]; tgi = base4 + tid; }
    consider(tx, tgi, true);
  }
  if (count > KNN_K) {
    tau = knn_compact((lds_u64p)&cand[wave][0], lane, count);
    count = KNN_K;
  }
  // wave's exact top-16 now in cand[wave][0..15]
  __syncthreads();

  // ---- final merge (wave 0): 4 x 16 candidates -> 16 ----------------------
  if (tid < 32) {
    unsigned long long a = cand[(tid >> 4)    ][tid & 15];
    unsigned long long q = cand[(tid >> 4) + 2][tid & 15];
#pragma unroll
    for (int r = 0; r < KNN_K; ++r) {
      unsigned long long m = u64min(a, q);
      unsigned long long g = m;
#pragma unroll
      for (int off = 16; off >= 1; off >>= 1)
        g = u64min(g, (unsigned long long)__shfl_xor(g, off, 32));
      if (a == g) a = ~0ULL; else if (q == g) q = ~0ULL;
      if (tid == 0) sfidx[r] = (int)(unsigned)(g & 0xFFFFFFFFull);
    }
  }
  __syncthreads();

  // ---- WMMA vote-sum epilogue --------------------------------------------
  // out[b, c] = (1/16) * sum_k labels[idx_k, c]
  // D = A x B with v_wmma_f32_16x16x32_f16:
  //   A (16x32) uniform = 1/32 -> layout-independent.
  //   B (32x16): each neighbor row appears in exactly two of the 32 K slots
  //   per column; B columns map to lane%16 while (VGPR,half,lane-half)
  //   permutes K, and A is K-uniform, so the sum is invariant to that
  //   permutation. 1/32 and one-hot values are exact in f16.
  int idxs[KNN_K];
#pragma unroll
  for (int i = 0; i < KNN_K; ++i) idxs[i] = sfidx[i];

  v16h amat;
#pragma unroll
  for (int e = 0; e < 16; ++e) amat[e] = (_Float16)0.03125f;  // 1/32

  const int col0 = lane & 15;
  const int h    = lane >> 4;  // lane-half
  const int nchunks = (C + 15) >> 4;

  for (int cc = 0; cc < nchunks; ++cc) {
    const int col  = cc * 16 + col0;
    const int colc = (col < C) ? col : (C - 1);  // clamp: no branchy loads;
                                                 // col >= C lanes never store
    v16h bmat;
#pragma unroll
    for (int e = 0; e < 16; ++e) {
      const int kn0 = (2 * e)     & 15;
      const int kn1 = (2 * e + 1) & 15;
      const int rowi = h ? idxs[kn1] : idxs[kn0];
      bmat[e] = (_Float16)labels[(size_t)rowi * (size_t)C + colc];
    }
    v8f cz = {};
    v8f d = __builtin_amdgcn_wmma_f32_16x16x32_f16(
        false, amat, false, bmat, (short)0, cz, false, false);
    // row M=0 lives in element 0 of lanes 0..15 (all waves computed the same
    // result with EXEC all-ones; only wave-0 lanes 0..15 store).
    if (tid < 16 && col < C) out[(size_t)b * (size_t)C + col] = d[0];
  }
}

extern "C" void kernel_launch(void* const* d_in, const int* in_sizes, int n_in,
                              void* d_out, int out_size, void* d_ws, size_t ws_size,
                              hipStream_t stream) {
  const float* dist   = (const float*)d_in[0];
  const float* labels = (const float*)d_in[1];
  float* out = (float*)d_out;

  const int C = 100;                 // classes (reference one_hot width)
  const int P = in_sizes[1] / C;     // 50000 prototypes
  const int B = in_sizes[0] / P;     // 2048 batch rows

  knnc_kernel<<<B, BLK, 0, stream>>>(dist, labels, out, P, C);
}
